// Transformer_PostLN_86844238725640
// MI455X (gfx1250) — compile-verified, matrix operand order checked by reading, not run on hardware
//
#include <hip/hip_runtime.h>

// ---------------------------------------------------------------- types
typedef __attribute__((ext_vector_type(16))) __bf16 v16bf;
typedef __attribute__((ext_vector_type(8)))  float  v8f;
typedef __attribute__((ext_vector_type(4)))  int    v4i;

// CDNA5 async global->LDS path (ASYNCcnt-tracked), with portable fallback.
#if defined(__AMDGCN__) && __has_builtin(__builtin_amdgcn_global_load_async_to_lds_b128) && __has_builtin(__builtin_amdgcn_s_wait_asynccnt)
#define HAVE_ASYNC 1
#else
#define HAVE_ASYNC 0
#endif

constexpr int B_   = 4;
constexpr int L_   = 2048;
constexpr int D_   = 1024;
constexpr int H_   = 16;
constexpr int DK_  = 64;
constexpr int DFF_ = 4096;
constexpr int M_   = B_ * L_;          // 8192 rows

__device__ __forceinline__ unsigned short f2bf(float f) {
  union { __bf16 h; unsigned short u; } x;
  x.h = (__bf16)f;
  return x.u;
}

// Copy 16 bf16 (32B) from global to LDS. Async (ASYNCcnt) path when available.
__device__ __forceinline__ void copy16(const unsigned short* g, unsigned short* l) {
#if HAVE_ASYNC
  __builtin_amdgcn_global_load_async_to_lds_b128((v4i*)g, (v4i*)l, 0, 0);
  __builtin_amdgcn_global_load_async_to_lds_b128((v4i*)(g + 8), (v4i*)(l + 8), 0, 0);
#else
  *(v4i*)l       = *(const v4i*)g;
  *(v4i*)(l + 8) = *(const v4i*)(g + 8);
#endif
}

__device__ __forceinline__ void async_fence() {
#if HAVE_ASYNC
  __builtin_amdgcn_s_wait_asynccnt(0);
#endif
}

// Load one 16x32 bf16 A/B fragment from an LDS row.
// rowbase points at the 32 contiguous bf16 of this lane's row (its k-chunk).
// sel = lane>>4 : lanes 0-15 hold K=[0,8)∪[16,24), lanes 16-31 K=[8,16)∪[24,32).
__device__ __forceinline__ v16bf load_frag(const unsigned short* rowbase, int sel) {
  union { v16bf v; v4i c[2]; } u;
  const v4i* p = (const v4i*)rowbase;
  u.c[0] = p[sel];
  u.c[1] = p[2 + sel];
  return u.v;
}

__device__ __forceinline__ v8f wmma_bf16(v16bf a, v16bf b, v8f c) {
  return __builtin_amdgcn_wmma_f32_16x16x32_bf16(false, a, false, b,
                                                 (short)0, c, false, false);
}

// ---------------------------------------------------------------- fp32 -> bf16
__global__ __launch_bounds__(256)
void cvt_f32_bf16(const float* __restrict__ in, unsigned short* __restrict__ out, int n) {
  int i = blockIdx.x * 256 + threadIdx.x;
  if (i < n) out[i] = f2bf(in[i]);
}

// ---------------------------------------------------------------- GEMM
// C[M,N] = A[M,K] * Bw[N,K]^T + bias, epilogue by MODE.
// MODE 0: bf16 scatter to [B,H,L,DK]      (Q, K projections)
// MODE 1: bf16 scatter to [B,H,DK,L]      (V projection, stored transposed)
// MODE 2: fp32  out = acc + bias + resid  (WO -> r ; FFN2 -> final)
// MODE 3: bf16  relu(acc + bias)          (FFN1)
//
// 128 threads = 4 waves; block tile 128x128, wave tile 64x64, BK=32,
// double-buffered LDS staged with async global->LDS copies.
constexpr int LDT = 40;  // padded LDS row stride in halves (80B, 16B aligned)

template<int MODE>
__global__ __launch_bounds__(128)
void gemm_bf16(const unsigned short* __restrict__ A,
               const unsigned short* __restrict__ Bw,
               const float* __restrict__ bias,
               const float* __restrict__ resid,
               void* __restrict__ Out,
               int M, int N, int K)
{
  __shared__ unsigned short As[2][128 * LDT];
  __shared__ unsigned short Bs[2][128 * LDT];

  const int tid  = threadIdx.x;            // 0..127
  const int lane = tid & 31;
  const int wave = tid >> 5;               // 0..3
  const int wm   = wave >> 1;              // 0..1  (64 rows)
  const int wn   = wave & 1;               // 0..1  (64 cols)
  const int bm   = blockIdx.y * 128;
  const int bn   = blockIdx.x * 128;
  const int g    = lane >> 4;
  const int cn   = lane & 15;

  v8f acc[4][4];
#pragma unroll
  for (int i = 0; i < 4; i++)
#pragma unroll
    for (int j = 0; j < 4; j++) acc[i][j] = (v8f){0,0,0,0,0,0,0,0};

  // each thread stages one full row (32 halves) of A and of B per k-step
  auto stage = [&](int buf, int kt) {
    const size_t ga = (size_t)(bm + tid) * K + kt;
    const size_t gb = (size_t)(bn + tid) * K + kt;
    copy16(&A[ga],       &As[buf][tid * LDT]);
    copy16(&A[ga + 16],  &As[buf][tid * LDT + 16]);
    copy16(&Bw[gb],      &Bs[buf][tid * LDT]);
    copy16(&Bw[gb + 16], &Bs[buf][tid * LDT + 16]);
  };

  stage(0, 0);
  async_fence();
  __syncthreads();

  const int nk = K >> 5;
  for (int it = 0; it < nk; ++it) {
    const int cur = it & 1;
    if (it + 1 < nk) stage(cur ^ 1, (it + 1) << 5);

    v16bf bf[4];
#pragma unroll
    for (int nf = 0; nf < 4; nf++)
      bf[nf] = load_frag(&Bs[cur][(wn * 64 + nf * 16 + cn) * LDT], g);
#pragma unroll
    for (int mf = 0; mf < 4; mf++) {
      v16bf af = load_frag(&As[cur][(wm * 64 + mf * 16 + cn) * LDT], g);
#pragma unroll
      for (int nf = 0; nf < 4; nf++)
        acc[mf][nf] = wmma_bf16(af, bf[nf], acc[mf][nf]);
    }
    async_fence();
    __syncthreads();
  }

#pragma unroll
  for (int mf = 0; mf < 4; mf++) {
#pragma unroll
    for (int nf = 0; nf < 4; nf++) {
#pragma unroll
      for (int j = 0; j < 8; j++) {
        const int row = bm + wm * 64 + mf * 16 + j + 8 * g;
        const int col = bn + wn * 64 + nf * 16 + cn;
        float v = acc[mf][nf][j] + bias[col];
        if (MODE == 0) {
          const int b = row >> 11, l = row & (L_ - 1);
          const int h = col >> 6,  dk = col & (DK_ - 1);
          ((unsigned short*)Out)[((((size_t)b * H_ + h) * L_ + l) << 6) + dk] = f2bf(v);
        } else if (MODE == 1) {
          const int b = row >> 11, l = row & (L_ - 1);
          const int h = col >> 6,  dk = col & (DK_ - 1);
          ((unsigned short*)Out)[(((size_t)b * H_ + h) * DK_ + dk) * L_ + l] = f2bf(v);
        } else if (MODE == 2) {
          const size_t o = (size_t)row * N + col;
          ((float*)Out)[o] = v + resid[o];
        } else {
          ((unsigned short*)Out)[(size_t)row * N + col] = f2bf(v > 0.f ? v : 0.f);
        }
      }
    }
  }
}

// ---------------------------------------------------------------- flash attention
// One block: 128 query rows of one (b,h). 8 waves, wave owns 16 q rows.
// Online softmax over 64-key chunks; scores never touch global memory.
constexpr int LDF = 72;  // padded stride for 64-wide tiles (144B)

__global__ __launch_bounds__(256)
void attn_flash(const unsigned short* __restrict__ q,    // [B,H,L,DK]
                const unsigned short* __restrict__ k,    // [B,H,L,DK]
                const unsigned short* __restrict__ vT,   // [B,H,DK,L]
                const int* __restrict__ maskPAD,         // [B,L,L]
                unsigned short* __restrict__ z)          // [B,L,H*DK]
{
  __shared__ unsigned short Qs[128 * LDF];
  __shared__ unsigned short Ks[64 * LDF];
  __shared__ unsigned short Vs[64 * LDF];
  __shared__ unsigned short Ps[8][16 * LDF];

  const int tid  = threadIdx.x;
  const int lane = tid & 31;
  const int wave = tid >> 5;
  const int g    = lane >> 4;
  const int cn   = lane & 15;
  const int qt   = blockIdx.x;   // 0..15
  const int h    = blockIdx.y;
  const int b    = blockIdx.z;

  const size_t headBase = ((size_t)b * H_ + h) * (size_t)L_ * DK_;

  { // stage Q tile (128x64)
    const int r = tid >> 2;
    const int c = (tid & 3) * 16;
    copy16(&q[headBase + (size_t)(qt * 128 + r) * DK_ + c],      &Qs[r * LDF + c]);
    copy16(&q[headBase + (size_t)(qt * 128 + r + 64) * DK_ + c], &Qs[(r + 64) * LDF + c]);
  }

  float m_run[8], l_run[8];
  v8f accO[4];
#pragma unroll
  for (int j = 0; j < 8; j++) { m_run[j] = -3.0e38f; l_run[j] = 0.f; }
#pragma unroll
  for (int nf = 0; nf < 4; nf++) accO[nf] = (v8f){0,0,0,0,0,0,0,0};

  for (int ck = 0; ck < L_ / 64; ck++) {
    __syncthreads();
    { // stage K chunk (64 keys x 64 dk) and V^T chunk (64 dk x 64 keys)
      const int r = tid >> 2;
      const int c = (tid & 3) * 16;
      copy16(&k[headBase + (size_t)(ck * 64 + r) * DK_ + c], &Ks[r * LDF + c]);
      copy16(&vT[headBase + (size_t)r * L_ + ck * 64 + c],   &Vs[r * LDF + c]);
    }
    async_fence();
    __syncthreads();

    // S = Q * K^T  (16 q-rows x 64 keys per wave)
    v8f s[4];
#pragma unroll
    for (int nf = 0; nf < 4; nf++) s[nf] = (v8f){0,0,0,0,0,0,0,0};
#pragma unroll
    for (int kk = 0; kk < 64; kk += 32) {
      v16bf aq = load_frag(&Qs[(wave * 16 + cn) * LDF + kk], g);
#pragma unroll
      for (int nf = 0; nf < 4; nf++) {
        v16bf bk = load_frag(&Ks[(nf * 16 + cn) * LDF + kk], g);
        s[nf] = wmma_bf16(aq, bk, s[nf]);
      }
    }

    // scale + mask + online softmax
    const float scale = 0.125f;  // 1/sqrt(64)
#pragma unroll
    for (int j = 0; j < 8; j++) {
      const int qrow = qt * 128 + wave * 16 + j + 8 * g;
      const int* mrow = &maskPAD[((size_t)b * L_ + qrow) * L_ + ck * 64];
      float sv[4];
#pragma unroll
      for (int nf = 0; nf < 4; nf++) {
        float x = s[nf][j] * scale;
        sv[nf] = (mrow[nf * 16 + cn] == 0) ? -32767.f : x;
      }
      float mx = fmaxf(fmaxf(sv[0], sv[1]), fmaxf(sv[2], sv[3]));
#pragma unroll
      for (int o = 8; o >= 1; o >>= 1) mx = fmaxf(mx, __shfl_xor(mx, o, 16));
      const float mnew = fmaxf(m_run[j], mx);
      const float fac  = __expf(m_run[j] - mnew);
      m_run[j] = mnew;
      float psum = 0.f;
#pragma unroll
      for (int nf = 0; nf < 4; nf++) {
        float p = __expf(sv[nf] - mnew);
        psum += p;
        Ps[wave][(j + 8 * g) * LDF + nf * 16 + cn] = f2bf(p);
      }
#pragma unroll
      for (int o = 8; o >= 1; o >>= 1) psum += __shfl_xor(psum, o, 16);
      l_run[j] = l_run[j] * fac + psum;
#pragma unroll
      for (int nf = 0; nf < 4; nf++) accO[nf][j] *= fac;
    }
    __syncthreads();

    // O += P * V
#pragma unroll
    for (int kk = 0; kk < 64; kk += 32) {
      v16bf ap = load_frag(&Ps[wave][cn * LDF + kk], g);
#pragma unroll
      for (int nf = 0; nf < 4; nf++) {
        v16bf bv = load_frag(&Vs[(nf * 16 + cn) * LDF + kk], g);
        accO[nf] = wmma_bf16(ap, bv, accO[nf]);
      }
    }
  }

  // epilogue: z[b, l, h*64 + dk] = O / l
#pragma unroll
  for (int j = 0; j < 8; j++) {
    const int qrow = qt * 128 + wave * 16 + j + 8 * g;
    const float inv = 1.f / l_run[j];
#pragma unroll
    for (int nf = 0; nf < 4; nf++) {
      z[((size_t)b * L_ + qrow) * (H_ * DK_) + h * DK_ + nf * 16 + cn] =
          f2bf(accO[nf][j] * inv);
    }
  }
}

// ---------------------------------------------------------------- layernorm
__global__ __launch_bounds__(256)
void layernorm_bf16(const float* __restrict__ r, const float* __restrict__ gam,
                    const float* __restrict__ bet, unsigned short* __restrict__ out)
{
  __shared__ float red[16];
  const int row = blockIdx.x;
  const float* x = r + (size_t)row * D_;
  float v[4], s = 0.f, ss = 0.f;
#pragma unroll
  for (int i = 0; i < 4; i++) {
    v[i] = x[threadIdx.x + i * 256];
    s += v[i]; ss += v[i] * v[i];
  }
#pragma unroll
  for (int o = 16; o >= 1; o >>= 1) { s += __shfl_xor(s, o, 32); ss += __shfl_xor(ss, o, 32); }
  const int wave = threadIdx.x >> 5, lane = threadIdx.x & 31;
  if (lane == 0) { red[wave] = s; red[8 + wave] = ss; }
  __syncthreads();
  if (wave == 0) {
    float a  = (lane < 8) ? red[lane] : 0.f;
    float bq = (lane < 8) ? red[8 + lane] : 0.f;
#pragma unroll
    for (int o = 4; o >= 1; o >>= 1) { a += __shfl_xor(a, o, 8); bq += __shfl_xor(bq, o, 8); }
    if (lane == 0) { red[0] = a; red[1] = bq; }
  }
  __syncthreads();
  const float mu   = red[0] / D_;
  const float var  = red[1] / D_ - mu * mu;
  const float rstd = rsqrtf(var + 1e-5f);
#pragma unroll
  for (int i = 0; i < 4; i++) {
    const int c = threadIdx.x + i * 256;
    out[(size_t)row * D_ + c] = f2bf((v[i] - mu) * rstd * gam[c] + bet[c]);
  }
}

// ---------------------------------------------------------------- launch
extern "C" void kernel_launch(void* const* d_in, const int* in_sizes, int n_in,
                              void* d_out, int out_size, void* d_ws, size_t ws_size,
                              hipStream_t stream) {
  (void)in_sizes; (void)n_in; (void)out_size; (void)ws_size;
  const float* x    = (const float*)d_in[0];
  const int*   mask = (const int*)d_in[1];
  const float* WQw  = (const float*)d_in[2];
  const float* WQb  = (const float*)d_in[3];
  const float* WKw  = (const float*)d_in[4];
  const float* WKb  = (const float*)d_in[5];
  const float* WVw  = (const float*)d_in[6];
  const float* WVb  = (const float*)d_in[7];
  const float* WOw  = (const float*)d_in[8];
  const float* WOb  = (const float*)d_in[9];
  const float* lng  = (const float*)d_in[10];
  const float* lnb  = (const float*)d_in[11];
  const float* W1w  = (const float*)d_in[12];
  const float* W1b  = (const float*)d_in[13];
  const float* W2w  = (const float*)d_in[14];
  const float* W2b  = (const float*)d_in[15];

  char* w = (char*)d_ws;
  size_t off = 0;
  auto alloc = [&](size_t bytes) {
    void* p = w + off;
    off += (bytes + 255) & ~(size_t)255;
    return p;
  };
  unsigned short* xb  = (unsigned short*)alloc((size_t)M_ * D_ * 2);
  unsigned short* wqb = (unsigned short*)alloc((size_t)D_ * D_ * 2);
  unsigned short* wkb = (unsigned short*)alloc((size_t)D_ * D_ * 2);
  unsigned short* wvb = (unsigned short*)alloc((size_t)D_ * D_ * 2);
  unsigned short* wob = (unsigned short*)alloc((size_t)D_ * D_ * 2);
  unsigned short* w1b = (unsigned short*)alloc((size_t)D_ * DFF_ * 2);
  unsigned short* w2b = (unsigned short*)alloc((size_t)D_ * DFF_ * 2);
  unsigned short* qb  = (unsigned short*)alloc((size_t)M_ * D_ * 2);
  unsigned short* kb  = (unsigned short*)alloc((size_t)M_ * D_ * 2);
  unsigned short* vtb = (unsigned short*)alloc((size_t)M_ * D_ * 2);
  unsigned short* zb  = (unsigned short*)alloc((size_t)M_ * D_ * 2);
  float*          rr  = (float*)alloc((size_t)M_ * D_ * 4);
  unsigned short* hln = (unsigned short*)alloc((size_t)M_ * D_ * 2);
  unsigned short* f1  = (unsigned short*)alloc((size_t)M_ * DFF_ * 2);

  // 1) fp32 -> bf16 conversions
  auto cvt = [&](const float* src, unsigned short* dst, int n) {
    cvt_f32_bf16<<<(n + 255) / 256, 256, 0, stream>>>(src, dst, n);
  };
  cvt(x,   xb,  M_ * D_);
  cvt(WQw, wqb, D_ * D_);
  cvt(WKw, wkb, D_ * D_);
  cvt(WVw, wvb, D_ * D_);
  cvt(WOw, wob, D_ * D_);
  cvt(W1w, w1b, D_ * DFF_);
  cvt(W2w, w2b, D_ * DFF_);

  // 2) Q/K/V projections
  dim3 gQKV(D_ / 128, M_ / 128);
  gemm_bf16<0><<<gQKV, 128, 0, stream>>>(xb, wqb, WQb, nullptr, qb,  M_, D_, D_);
  gemm_bf16<0><<<gQKV, 128, 0, stream>>>(xb, wkb, WKb, nullptr, kb,  M_, D_, D_);
  gemm_bf16<1><<<gQKV, 128, 0, stream>>>(xb, wvb, WVb, nullptr, vtb, M_, D_, D_);

  // 3) fused attention (flash, online softmax)
  attn_flash<<<dim3(L_ / 128, H_, B_), 256, 0, stream>>>(qb, kb, vtb, mask, zb);

  // 4) output projection + residual 1:  r = x + z*WO^T + b
  gemm_bf16<2><<<dim3(D_ / 128, M_ / 128), 128, 0, stream>>>(zb, wob, WOb, x, rr, M_, D_, D_);

  // 5) pre-LN for FFN
  layernorm_bf16<<<M_, 256, 0, stream>>>(rr, lng, lnb, hln);

  // 6) FFN1 (relu)
  gemm_bf16<3><<<dim3(DFF_ / 128, M_ / 128), 128, 0, stream>>>(hln, w1b, W1b, nullptr, f1, M_, DFF_, D_);

  // 7) FFN2 + residual 2 -> final output
  gemm_bf16<2><<<dim3(D_ / 128, M_ / 128), 128, 0, stream>>>(f1, w2b, W2b, rr, (float*)d_out, M_, D_, DFF_);
}
